// GIN_24223615549809
// MI455X (gfx1250) — matrix operand order repaired and checked by reading, not compiled, not run on hardware
//
#include <hip/hip_runtime.h>
#include <hip/hip_bf16.h>

typedef _Float16 v16h __attribute__((ext_vector_type(16)));
typedef float    v8f  __attribute__((ext_vector_type(8)));

#define FEAT 64
#define EPS 0.1f
#define BN_EPS 1e-5f

// ---------------------------------------------------------------------------
// Pre-swizzle W [64x64 f32 row-major, K x N] into f16 WMMA B-fragment order.
// Fragment buffer layout: Wh[((nb*2 + kc)*32 + lane)*16 + j]
//   nb   = 16-column tile (0..3), kc = 32-wide K chunk (0..1)
//   lane = wave lane (0..31), j = fragment half index (0..15)
// B layout (16-bit, 32x16): lane L holds column N = L&15,
//   K = (L>>4)*8 + j (j<8), 16 + (L>>4)*8 + (j-8) (j>=8) within the 32-chunk.
// ---------------------------------------------------------------------------
__global__ void gin_prep_w_kernel(const float* __restrict__ W,
                                  _Float16* __restrict__ Wh) {
    int tid = blockIdx.x * blockDim.x + threadIdx.x;   // 0..4095
    if (tid >= 4096) return;
    int j    = tid & 15;
    int lane = (tid >> 4) & 31;
    int kc   = (tid >> 9) & 1;
    int nb   = tid >> 10;
    int half = lane >> 4;
    int n    = nb * 16 + (lane & 15);
    int k    = kc * 32 + half * 8 + (j & 7) + ((j >> 3) << 4);
    Wh[tid] = (_Float16)W[k * FEAT + n];
}

// Agg = (1 + eps) * H   (full overwrite; also clears ws poison)
__global__ void gin_init_agg_kernel(const float* __restrict__ Hin,
                                    float* __restrict__ Agg, int elem4) {
    int i = blockIdx.x * blockDim.x + threadIdx.x;
    if (i >= elem4) return;
    float4 v = ((const float4*)Hin)[i];
    v.x *= (1.0f + EPS); v.y *= (1.0f + EPS);
    v.z *= (1.0f + EPS); v.w *= (1.0f + EPS);
    ((float4*)Agg)[i] = v;
}

// Edge scatter-add: 16 threads per edge, 4 floats each (coalesced row read,
// L2-resident f32 atomic RMWs).
__global__ void gin_scatter_kernel(const float* __restrict__ Hsrc,
                                   const int* __restrict__ src,
                                   const int* __restrict__ dst,
                                   float* __restrict__ Agg, int nEdges) {
    int t = blockIdx.x * blockDim.x + threadIdx.x;
    int e = t >> 4;
    if (e >= nEdges) return;
    int c4 = (t & 15) * 4;
    int s = src[e], d = dst[e];
    float4 v = *(const float4*)(Hsrc + (size_t)s * FEAT + c4);
    float* o = Agg + (size_t)d * FEAT + c4;
    atomicAdd(o + 0, v.x);
    atomicAdd(o + 1, v.y);
    atomicAdd(o + 2, v.z);
    atomicAdd(o + 3, v.w);
}

// ---------------------------------------------------------------------------
// H = relu(bn(Agg @ W)).  One wave per 16x16 output tile; 4 waves per block
// cover all 64 columns of one 16-row stripe. K=64 -> 2x wmma_f32_16x16x32_f16.
// A layout (16-bit 16x32): lane L holds row M=L&15,
//   K = {0..7} u {16..23} shifted by +8 for lanes 16..31.
// Epilogue bounds check is UNIFORM per block (scalar branch, no EXEC churn);
// the full-tile fast path stores with immediate offsets.
// ---------------------------------------------------------------------------
__global__ void gin_gemm_bn_relu_kernel(const float* __restrict__ A,
                                        const _Float16* __restrict__ Wh,
                                        const float* __restrict__ bn_scale,
                                        const float* __restrict__ bn_bias,
                                        const float* __restrict__ bn_mean,
                                        const float* __restrict__ bn_var,
                                        float* __restrict__ H, int nNodes) {
    const int lane = threadIdx.x & 31;
    const int nb   = threadIdx.x >> 5;          // column tile 0..3
    const int half = lane >> 4;
    const int r0   = blockIdx.x << 4;
    int mrow = r0 + (lane & 15);
    if (mrow >= nNodes) mrow = nNodes - 1;      // uniform-safe clamp (no EXEC holes)
    const int n = (nb << 4) + (lane & 15);

    const float* arow = A + (size_t)mrow * FEAT + half * 8;

    v8f c = {};
#pragma unroll
    for (int kc = 0; kc < 2; ++kc) {
        const float4 f0 = *(const float4*)(arow + kc * 32);
        const float4 f1 = *(const float4*)(arow + kc * 32 + 4);
        const float4 f2 = *(const float4*)(arow + kc * 32 + 16);
        const float4 f3 = *(const float4*)(arow + kc * 32 + 20);
        v16h a;
        a[0]  = (_Float16)f0.x; a[1]  = (_Float16)f0.y;
        a[2]  = (_Float16)f0.z; a[3]  = (_Float16)f0.w;
        a[4]  = (_Float16)f1.x; a[5]  = (_Float16)f1.y;
        a[6]  = (_Float16)f1.z; a[7]  = (_Float16)f1.w;
        a[8]  = (_Float16)f2.x; a[9]  = (_Float16)f2.y;
        a[10] = (_Float16)f2.z; a[11] = (_Float16)f2.w;
        a[12] = (_Float16)f3.x; a[13] = (_Float16)f3.y;
        a[14] = (_Float16)f3.z; a[15] = (_Float16)f3.w;
        const v16h b = *(const v16h*)(Wh + ((((nb << 1) + kc) << 5) | lane) * 16);
        c = __builtin_amdgcn_wmma_f32_16x16x32_f16(
                /*neg_a=*/false, a, /*neg_b=*/false, b,
                /*c_mod=*/(short)0, c, /*reuse_a=*/false, /*reuse_b=*/false);
    }

    // fused inference BatchNorm + ReLU epilogue (per-column params)
    const float istd = rsqrtf(bn_var[n] + BN_EPS);
    const float sc = bn_scale[n] * istd;
    const float bi = bn_bias[n] - bn_mean[n] * sc;
    const int mbase = r0 + half * 8;
    float* __restrict__ hp = H + (size_t)mbase * FEAT + n;

    if (r0 + 16 <= nNodes) {
        // full tile: uniform scalar branch, unguarded stores, imm offsets
#pragma unroll
        for (int r = 0; r < 8; ++r) {
            float val = c[r] * sc + bi;
            hp[r * FEAT] = val > 0.0f ? val : 0.0f;
        }
    } else {
        // ragged tail tile (not taken for N % 16 == 0)
#pragma unroll
        for (int r = 0; r < 8; ++r) {
            if (mbase + r < nNodes) {
                float val = c[r] * sc + bi;
                hp[r * FEAT] = val > 0.0f ? val : 0.0f;
            }
        }
    }
}

// Sum-pool over nodes: per-thread register partials (fixed column since the
// grid stride is a multiple of 64), LDS combine, few global atomics per block.
__global__ void gin_pool_kernel(const float* __restrict__ H,
                                float* __restrict__ pooled, int total) {
    __shared__ float s[FEAT];
    if (threadIdx.x < FEAT) s[threadIdx.x] = 0.0f;
    __syncthreads();
    int idx = blockIdx.x * blockDim.x + threadIdx.x;
    int stride = gridDim.x * blockDim.x;      // multiple of 64 by construction
    float acc = 0.0f;
    for (int i = idx; i < total; i += stride) acc += H[i];
    atomicAdd(&s[idx & (FEAT - 1)], acc);
    __syncthreads();
    if (threadIdx.x < FEAT) atomicAdd(&pooled[threadIdx.x], s[threadIdx.x]);
}

__global__ void gin_classify_kernel(const float* __restrict__ pooled,
                                    const float* __restrict__ Wout,
                                    float* __restrict__ out, int nClasses) {
    __shared__ float p[FEAT];
    if (threadIdx.x < FEAT) p[threadIdx.x] = pooled[threadIdx.x];
    __syncthreads();
    int ccls = threadIdx.x;
    if (ccls < nClasses) {
        float acc = 0.0f;
#pragma unroll
        for (int k = 0; k < FEAT; ++k) acc += p[k] * Wout[k * nClasses + ccls];
        out[ccls] = acc;
    }
}

extern "C" void kernel_launch(void* const* d_in, const int* in_sizes, int n_in,
                              void* d_out, int out_size, void* d_ws, size_t ws_size,
                              hipStream_t stream) {
    const float* x        = (const float*)d_in[0];
    const int*   src      = (const int*)d_in[1];
    const int*   dst      = (const int*)d_in[2];
    const float* W0       = (const float*)d_in[3];
    const float* W1       = (const float*)d_in[4];
    const float* bn_scale = (const float*)d_in[5];
    const float* bn_bias  = (const float*)d_in[6];
    const float* bn_mean  = (const float*)d_in[7];
    const float* bn_var   = (const float*)d_in[8];
    const float* W_out    = (const float*)d_in[9];
    float*       out      = (float*)d_out;

    const int nNodes   = in_sizes[0] / FEAT;   // 50000
    const int nEdges   = in_sizes[1];          // 800000
    const int nClasses = out_size;             // 100

    // workspace layout (hipMalloc base is 256B aligned; offsets stay aligned)
    char* ws = (char*)d_ws;
    const size_t bytesMat = (size_t)nNodes * FEAT * sizeof(float); // 12.8 MB
    float*     Agg    = (float*)(ws);
    float*     Hbuf   = (float*)(ws + bytesMat);
    _Float16*  Wh0    = (_Float16*)(ws + 2 * bytesMat);
    _Float16*  Wh1    = Wh0 + 4096;
    float*     pooled = (float*)(ws + 2 * bytesMat + 16384);

    const int elem4 = nNodes * FEAT / 4;
    const int tiles = (nNodes + 15) / 16;                 // 3125
    const int sBlk  = (nEdges * 16 + 255) / 256;          // scatter blocks
    const int iBlk  = (elem4 + 255) / 256;

    // weight pre-swizzle + pooled clear
    gin_prep_w_kernel<<<8, 512, 0, stream>>>(W0, Wh0);
    gin_prep_w_kernel<<<8, 512, 0, stream>>>(W1, Wh1);
    hipMemsetAsync(pooled, 0, FEAT * sizeof(float), stream);

    // layer 0: Agg = (A + (1+eps)I) x ; H = relu(bn(Agg @ W0))
    gin_init_agg_kernel<<<iBlk, 256, 0, stream>>>(x, Agg, elem4);
    gin_scatter_kernel<<<sBlk, 256, 0, stream>>>(x, src, dst, Agg, nEdges);
    gin_gemm_bn_relu_kernel<<<tiles, 128, 0, stream>>>(
        Agg, Wh0, bn_scale, bn_bias, bn_mean, bn_var, Hbuf, nNodes);

    // layer 1
    gin_init_agg_kernel<<<iBlk, 256, 0, stream>>>(Hbuf, Agg, elem4);
    gin_scatter_kernel<<<sBlk, 256, 0, stream>>>(Hbuf, src, dst, Agg, nEdges);
    gin_gemm_bn_relu_kernel<<<tiles, 128, 0, stream>>>(
        Agg, Wh1, bn_scale, bn_bias, bn_mean, bn_var, Hbuf, nNodes);

    // pool + classify
    gin_pool_kernel<<<512, 256, 0, stream>>>(Hbuf, pooled, nNodes * FEAT);
    gin_classify_kernel<<<1, 128, 0, stream>>>(pooled, W_out, out, nClasses);
}
